// GCN_Embed_20383914787352
// MI455X (gfx1250) — matrix operand order repaired and checked by reading, not compiled, not run on hardware
//
#include <hip/hip_runtime.h>

// ---------------------------------------------------------------------------
// Types
// ---------------------------------------------------------------------------
typedef _Float16 h16;
typedef __attribute__((ext_vector_type(16))) _Float16 v16h;
typedef __attribute__((ext_vector_type(8)))  _Float16 v8h;
typedef __attribute__((ext_vector_type(8)))  float    v8f;

#define F_BIAS 1
#define F_BN   2
#define F_RES  4
#define F_RELU 8

// ---------------------------------------------------------------------------
// Structural constants (from reference)
// ---------------------------------------------------------------------------
static const int BB  = 16;        // batch
static const int SS  = 256;       // sequence
static const int NN  = BB * SS;   // 4096 per-frame graphs
static const int KK  = 3;         // adjacency kernel size
static const int CIN = 3;
static const int KPAD = 32;       // layer-0 K padded to one WMMA K-step
static const int FC  = 512;
static const int NC  = 1024;      // frame chunk for GCN stages (keeps M % 128 == 0)

// Graph partitions baked into constant memory (no memcpy -> capture safe)
__constant__ int c_poff[2][8] = {
  {0, 6, 9, 12, 15, 18, 21, 0},   // hand: 6 parts
  {0, 3, 5, 7, 10, 13, 0, 0}      // body: 5 parts
};
__constant__ int c_pidx[2][21] = {
  {0,1,5,9,13,17, 2,3,4, 6,7,8, 10,11,12, 14,15,16, 18,19,20},
  {0,1,2, 3,4, 5,6, 7,8,9, 10,11,12, 0,0,0,0,0,0,0,0}
};

// ---------------------------------------------------------------------------
// Guard-free WMMA GEMM:  C = act(BN(A*B^T + bias) + res)
// REQUIRES: M % 128 == 0, N % 128 == 0, K % 32 == 0 (caller pads).
//   A: (M x K) f16 row-major, lda (halves, multiple of 32)
//   Bw: (N x K) f16 row-major (weights: each output column contiguous)
// Block 128x128, 8 waves (wave32), wave tile 32x64 = 2x4 wmma 16x16 frags.
// 16-bit fragment k order used here (contiguous, same for A and B):
//   A: lane L -> row m = L%16; halves = K(k0 + h*16 + 0..15); h=L/16
//   B: lane L -> col n = L%16; halves = K(k0 + h*16 + 0..15)
// ---------------------------------------------------------------------------
template <int FLAGS>
__global__ __launch_bounds__(256)
void gemm_wmma_t(const h16* __restrict__ A, int lda, int K,
                 const h16* __restrict__ Bw, int ldb,
                 float* __restrict__ Cf, h16* __restrict__ Ch, int ldc,
                 const float* __restrict__ bias,
                 const float* __restrict__ bnG, const float* __restrict__ bnB,
                 const float* __restrict__ bnM, const float* __restrict__ bnV,
                 const h16* __restrict__ res, int ldres) {
  const int lane = threadIdx.x & 31;
  const int wave = threadIdx.x >> 5;
  const int wm = wave & 3;
  const int wn = wave >> 2;
  const int bm = blockIdx.x * 128 + wm * 32;
  const int bn = blockIdx.y * 128 + wn * 64;
  const int ml = lane & 15;
  const int hh = lane >> 4;

  // Base pointers (lane-resolved), advanced by k0 each step.
  const h16* pa0 = A + (long)(bm + ml) * lda + hh * 16;        // A frag mt=0
  const h16* pa1 = A + (long)(bm + 16 + ml) * lda + hh * 16;   // A frag mt=1
  const h16* pb0 = Bw + (long)(bn +  0 + ml) * ldb + hh * 16;
  const h16* pb1 = Bw + (long)(bn + 16 + ml) * ldb + hh * 16;
  const h16* pb2 = Bw + (long)(bn + 32 + ml) * ldb + hh * 16;
  const h16* pb3 = Bw + (long)(bn + 48 + ml) * ldb + hh * 16;

  v8f acc[2][4];
#pragma unroll
  for (int a = 0; a < 2; ++a)
#pragma unroll
    for (int b = 0; b < 4; ++b)
#pragma unroll
      for (int r = 0; r < 8; ++r) acc[a][b][r] = 0.f;

  for (int k0 = 0; k0 < K; k0 += 32) {
    // A fragments: two 16B loads each
    v16h af[2];
    {
      v8h lo = *(const v8h*)(pa0 + k0);
      v8h hi = *(const v8h*)(pa0 + k0 + 8);
#pragma unroll
      for (int j = 0; j < 8; ++j) { af[0][j] = lo[j]; af[0][8 + j] = hi[j]; }
    }
    {
      v8h lo = *(const v8h*)(pa1 + k0);
      v8h hi = *(const v8h*)(pa1 + k0 + 8);
#pragma unroll
      for (int j = 0; j < 8; ++j) { af[1][j] = lo[j]; af[1][8 + j] = hi[j]; }
    }
    // Speculative prefetch of the next K tile of the A stream
    __builtin_prefetch(pa0 + k0 + 32, 0, 1);
    __builtin_prefetch(pa1 + k0 + 32, 0, 1);
    // B fragments: one 32B (aligned) load each
    v16h bf[4];
    bf[0] = *(const v16h*)(pb0 + k0);
    bf[1] = *(const v16h*)(pb1 + k0);
    bf[2] = *(const v16h*)(pb2 + k0);
    bf[3] = *(const v16h*)(pb3 + k0);
#pragma unroll
    for (int mt = 0; mt < 2; ++mt)
#pragma unroll
      for (int nt = 0; nt < 4; ++nt)
        acc[mt][nt] = __builtin_amdgcn_wmma_f32_16x16x32_f16(
            false, af[mt], false, bf[nt], (short)0, acc[mt][nt], false, false);
  }

  // Epilogue. D layout: lane col = lane%16, VGPR r -> row (lane/16)*8 + r.
#pragma unroll
  for (int nt = 0; nt < 4; ++nt) {
    int col = bn + nt * 16 + ml;
    float bi = (FLAGS & F_BIAS) ? bias[col] : 0.f;
    float sc = 1.f, sh = 0.f;
    if (FLAGS & F_BN) {
      sc = bnG[col] * rsqrtf(bnV[col] + 1e-5f);
      sh = bnB[col] - bnM[col] * sc;
    }
#pragma unroll
    for (int mt = 0; mt < 2; ++mt) {
#pragma unroll
      for (int r = 0; r < 8; ++r) {
        int row = bm + mt * 16 + hh * 8 + r;
        float v = acc[mt][nt][r] + bi;
        if (FLAGS & F_BN)  v = v * sc + sh;
        if (FLAGS & F_RES) v += (float)res[(long)row * ldres + col];
        if (FLAGS & F_RELU) v = fmaxf(v, 0.f);
        long o = (long)row * ldc + col;
        if (Cf) Cf[o] = v;
        if (Ch) Ch[o] = (h16)v;
      }
    }
  }
}

// ---------------------------------------------------------------------------
// Elementwise / small kernels
// ---------------------------------------------------------------------------
__global__ void cvt_f32_f16_k(const float* __restrict__ w, h16* __restrict__ o, long n) {
  long t = (long)blockIdx.x * blockDim.x + threadIdx.x;
  if (t < n) o[t] = (h16)w[t];
}

// f32 (rows x cols) -> f16 (rows x ldp) with zero padding of cols..ldp-1
__global__ void cvt_pad_k(const float* __restrict__ w, h16* __restrict__ o,
                          long rows, int cols, int ldp) {
  long t = (long)blockIdx.x * blockDim.x + threadIdx.x;
  long tot = rows * ldp;
  if (t >= tot) return;
  int c = (int)(t % ldp);
  long r = t / ldp;
  o[t] = (c < cols) ? (h16)w[r * cols + c] : (h16)0.f;
}

// data_bn over V*C features (eps = 0.1, reference quirk), f32 -> f16 rows
// (n*V+v, :) with row stride ldp (=32), cols C valid, rest zero.
__global__ void data_bn_k(const float* __restrict__ x, h16* __restrict__ X0,
                          long rows, int V, int C, int ldp,
                          const float* __restrict__ g, const float* __restrict__ b,
                          const float* __restrict__ m, const float* __restrict__ v) {
  long t = (long)blockIdx.x * blockDim.x + threadIdx.x;
  long tot = rows * ldp;
  if (t >= tot) return;
  int c = (int)(t % ldp);
  long r = t / ldp;
  if (c < C) {
    int f = (int)(r % V) * C + c;
    float sc = g[f] * rsqrtf(v[f] + 0.1f);
    X0[t] = (h16)((x[r * C + c] - m[f]) * sc + b[f]);
  } else {
    X0[t] = (h16)0.f;
  }
}

// Y2[(n,w),c] = sum_k sum_v Y1[(n,v), k*Cout+c] * A[k,v,w], fused BN1 + ReLU -> f16
__global__ void graph_gather_k(const h16* __restrict__ Y1, const float* __restrict__ A,
                               h16* __restrict__ H, int nF, int V, int Cout, int Kk,
                               const float* __restrict__ g, const float* __restrict__ b,
                               const float* __restrict__ m, const float* __restrict__ vv) {
  long t = (long)blockIdx.x * blockDim.x + threadIdx.x;
  long tot = (long)nF * V * Cout;
  if (t >= tot) return;
  int c = (int)(t % Cout);
  int w = (int)((t / Cout) % V);
  long n = t / ((long)Cout * V);
  const int KC = Kk * Cout;
  const h16* y = Y1 + n * (long)V * KC;
  float acc = 0.f;
  for (int k = 0; k < Kk; ++k) {
    const float* Ak = A + (long)k * V * V;
    const h16* yk = y + k * Cout + c;
    for (int q = 0; q < V; ++q)
      acc += (float)yk[(long)q * KC] * Ak[q * V + w];
  }
  float sc = g[c] * rsqrtf(vv[c] + 1e-5f);
  acc = (acc - m[c]) * sc + b[c];
  H[(n * V + w) * (long)Cout + c] = (h16)fmaxf(acc, 0.f);
}

// Graph-part max pool: Xp[(n,p),c] = max_{v in part p} X[(n,v),c]
__global__ void part_pool_k(const h16* __restrict__ X, h16* __restrict__ Xp,
                            int nF, int V, int C, int P, int set) {
  long t = (long)blockIdx.x * blockDim.x + threadIdx.x;
  long tot = (long)nF * P * C;
  if (t >= tot) return;
  int c = (int)(t % C);
  int p = (int)((t / C) % P);
  long n = t / ((long)C * P);
  int s = c_poff[set][p], e = c_poff[set][p + 1];
  float mx = -3.0e38f;
  for (int j = s; j < e; ++j) {
    float v = (float)X[(n * V + c_pidx[set][j]) * (long)C + c];
    mx = fmaxf(mx, v);
  }
  Xp[(n * P + p) * (long)C + c] = (h16)mx;
}

// Global graph max pool: Z[n,c] = max_p Xp[(n,p),c]
__global__ void gmax_k(const h16* __restrict__ Xp, h16* __restrict__ Z,
                       int nF, int P, int C) {
  long t = (long)blockIdx.x * blockDim.x + threadIdx.x;
  long tot = (long)nF * C;
  if (t >= tot) return;
  int c = (int)(t % C);
  long n = t / C;
  float mx = -3.0e38f;
  for (int p = 0; p < P; ++p)
    mx = fmaxf(mx, (float)Xp[(n * P + p) * (long)C + c]);
  Z[n * (long)C + c] = (h16)mx;
}

// im2col along S (same padding): out[n, i*T + t] = Z[b, s+t-pad, i] or 0
__global__ void im2col_k(const h16* __restrict__ Z, h16* __restrict__ out,
                         int Bq, int S, int C, int T) {
  long t = (long)blockIdx.x * blockDim.x + threadIdx.x;
  long tot = (long)Bq * S * C;
  if (t >= tot) return;
  int i = (int)(t % C);
  long n = t / C;
  int s = (int)(n % S);
  long b = n / S;
  int pad = T >> 1;
  h16* o = out + n * (long)(C * T) + (long)i * T;
  for (int tt = 0; tt < T; ++tt) {
    int s2 = s + tt - pad;
    o[tt] = (s2 >= 0 && s2 < S) ? Z[(b * S + s2) * (long)C + i] : (h16)0.f;
  }
}

// ---------------------------------------------------------------------------
// Host-side param indexing (pytree leaves in setup_inputs insertion order)
// ---------------------------------------------------------------------------
struct BNIdx { int g, b, m, v; };
struct LayerIdx { int gw, gb; BNIdx bn1; int tw, tb; BNIdx bn2; int rw, rb; BNIdx rbn; int mode; };
struct TcnIdx { int w5, b5, w3a, b3a, w3b, b3b, w1, b1; BNIdx bn; };
struct ModelIdx { BNIdx dbn; LayerIdx gcn[3]; LayerIdx pool[2]; int c4w, c4b; BNIdx c4bn; TcnIdx t1, t2; };

static void walkBN(int& i, BNIdx& b) { b.g = i++; b.b = i++; b.m = i++; b.v = i++; }
static void walkLayer(int& i, LayerIdx& l, int mode) {
  l.gw = i++; l.gb = i++; walkBN(i, l.bn1);
  l.tw = i++; l.tb = i++; walkBN(i, l.bn2);
  l.mode = mode; l.rw = -1; l.rb = -1;
  if (mode == 2) { l.rw = i++; l.rb = i++; walkBN(i, l.rbn); }
}
static void walkTcn(int& i, TcnIdx& t) {
  t.w5 = i++; t.b5 = i++; t.w3a = i++; t.b3a = i++; t.w3b = i++; t.b3b = i++;
  t.w1 = i++; t.b1 = i++; walkBN(i, t.bn);
}
static void walkModel(int& i, ModelIdx& m) {
  walkBN(i, m.dbn);
  walkLayer(i, m.gcn[0], 0); walkLayer(i, m.gcn[1], 1); walkLayer(i, m.gcn[2], 2);
  walkLayer(i, m.pool[0], 2); walkLayer(i, m.pool[1], 1);
  m.c4w = i++; m.c4b = i++; walkBN(i, m.c4bn);
  walkTcn(i, m.t1); walkTcn(i, m.t2);
}

// ---------------------------------------------------------------------------
// Host-side model pointer tables + workspace bump allocator
// ---------------------------------------------------------------------------
struct Bump {
  char* p; size_t off; size_t cap;
  void* get(size_t b) { void* r = p + off; off += (b + 255) & ~(size_t)255; return r; }
};

struct BNW { const float *g, *b, *m, *v; };
struct LayerW {
  const h16 *gw, *tw, *rw; const float *gb, *tb, *rb;
  BNW bn1, bn2, rbn; int mode, cin, cout, kd;  // kd = padded K for gcn/res GEMMs
};
struct TcnW { const h16 *w5, *w3a, *w3b, *w1; const float *b5, *b3a, *b3b, *b1; BNW bn; };
struct ModelW { BNW dbn; LayerW gcn[3]; LayerW pool[2]; const h16* c4w; const float* c4b; BNW c4bn; TcnW t1, t2; };

struct Bufs { h16 *X0, *Xa, *Xb, *Y1, *H, *R, *Zg, *Z1, *Z2, *I5, *I3, *Btmp, *Cat; };

static h16* cvtW(Bump& ws, hipStream_t st, const float* w, long n) {
  h16* o = (h16*)ws.get((size_t)n * sizeof(h16));
  cvt_f32_f16_k<<<(int)((n + 255) / 256), 256, 0, st>>>(w, o, n);
  return o;
}
static h16* cvtWPad(Bump& ws, hipStream_t st, const float* w, long rows, int cols, int ldp) {
  long n = rows * ldp;
  h16* o = (h16*)ws.get((size_t)n * sizeof(h16));
  cvt_pad_k<<<(int)((n + 255) / 256), 256, 0, st>>>(w, o, rows, cols, ldp);
  return o;
}

// Dispatch over the 5 epilogue combinations actually used.
static void gemm(hipStream_t st, const h16* A, int lda, int M, int Kd,
                 const h16* Bw, int ldb, int Nd,
                 float* Cf, h16* Ch, int ldc,
                 const float* bias, const BNW* bn, const h16* res, int ldres, bool relu) {
  int flags = 0;
  if (bias) flags |= F_BIAS;
  if (bn)   flags |= F_BN;
  if (res)  flags |= F_RES;
  if (relu) flags |= F_RELU;
  dim3 grid((unsigned)(M / 128), (unsigned)(Nd / 128));
  dim3 blk(256);
  const float* g = bn ? bn->g : nullptr;
  const float* b = bn ? bn->b : nullptr;
  const float* m = bn ? bn->m : nullptr;
  const float* v = bn ? bn->v : nullptr;
#define GEMM_CASE(F) \
  gemm_wmma_t<F><<<grid, blk, 0, st>>>(A, lda, Kd, Bw, ldb, Cf, Ch, ldc, bias, g, b, m, v, res, ldres)
  switch (flags) {
    case F_BIAS:                              GEMM_CASE(F_BIAS); break;
    case F_BIAS | F_BN:                       GEMM_CASE(F_BIAS | F_BN); break;
    case F_BIAS | F_RELU:                     GEMM_CASE(F_BIAS | F_RELU); break;
    case F_BIAS | F_BN | F_RELU:              GEMM_CASE(F_BIAS | F_BN | F_RELU); break;
    default:                                  GEMM_CASE(F_BIAS | F_BN | F_RES | F_RELU); break;
  }
#undef GEMM_CASE
}

static ModelW buildModel(Bump& ws, hipStream_t st, void* const* din, const ModelIdx& mi) {
  auto F = [&](int i) -> const float* { return (const float*)din[i]; };
  auto bn = [&](const BNIdx& b) { BNW r; r.g = F(b.g); r.b = F(b.b); r.m = F(b.m); r.v = F(b.v); return r; };
  ModelW w{};
  w.dbn = bn(mi.dbn);
  int cins[5]  = {CIN, 128, 128, 256, 512};
  int couts[5] = {128, 128, 256, 512, 512};
  for (int l = 0; l < 5; ++l) {
    const LayerIdx& li = (l < 3) ? mi.gcn[l] : mi.pool[l - 3];
    LayerW& lw = (l < 3) ? w.gcn[l] : w.pool[l - 3];
    lw.cin = cins[l]; lw.cout = couts[l]; lw.mode = li.mode;
    lw.kd = (lw.cin == CIN) ? KPAD : lw.cin;   // pad K=3 -> 32
    if (lw.cin == CIN)
      lw.gw = cvtWPad(ws, st, F(li.gw), (long)KK * lw.cout, lw.cin, KPAD);
    else
      lw.gw = cvtW(ws, st, F(li.gw), (long)KK * lw.cout * lw.cin);
    lw.gb = F(li.gb);
    lw.bn1 = bn(li.bn1);
    lw.tw = cvtW(ws, st, F(li.tw), (long)lw.cout * lw.cout); lw.tb = F(li.tb);
    lw.bn2 = bn(li.bn2);
    lw.rw = nullptr; lw.rb = nullptr;
    if (li.mode == 2) {
      lw.rw = cvtW(ws, st, F(li.rw), (long)lw.cout * lw.cin);
      lw.rb = F(li.rb); lw.rbn = bn(li.rbn);
    }
  }
  w.c4w = cvtW(ws, st, F(mi.c4w), (long)FC * FC); w.c4b = F(mi.c4b); w.c4bn = bn(mi.c4bn);
  auto tc = [&](const TcnIdx& t) {
    TcnW r;
    r.w5  = cvtW(ws, st, F(t.w5),  (long)FC * FC * 5); r.b5  = F(t.b5);
    r.w3a = cvtW(ws, st, F(t.w3a), (long)FC * FC * 3); r.b3a = F(t.b3a);
    r.w3b = cvtW(ws, st, F(t.w3b), (long)FC * FC * 3); r.b3b = F(t.b3b);
    r.w1  = cvtW(ws, st, F(t.w1),  (long)FC * 2 * FC); r.b1  = F(t.b1);
    r.bn = bn(t.bn); return r;
  };
  w.t1 = tc(mi.t1); w.t2 = tc(mi.t2);
  return w;
}

// One st_gcn layer on nF frames of a V-node graph: Xin(f16, ld=kd) -> Xout(f16, ld=cout)
static void run_layer(hipStream_t st, const LayerW& L, const float* Aadj,
                      int nF, int V, const h16* Xin, Bufs& bf, h16* Xout) {
  int Mc = nF * V;
  int KC = KK * L.cout;
  // gcn 1x1 conv to K*Cout channels (bias only)
  gemm(st, Xin, L.kd, Mc, L.kd, L.gw, L.kd, KC, nullptr, bf.Y1, KC,
       L.gb, nullptr, nullptr, 0, false);
  // adjacency einsum fused with BN1 + ReLU
  long tg = (long)nF * V * L.cout;
  graph_gather_k<<<(int)((tg + 255) / 256), 256, 0, st>>>(
      bf.Y1, Aadj, bf.H, nF, V, L.cout, KK, L.bn1.g, L.bn1.b, L.bn1.m, L.bn1.v);
  // residual
  const h16* res = nullptr;
  if (L.mode == 1) {
    res = Xin;  // identity (cin == cout)
  } else if (L.mode == 2) {
    gemm(st, Xin, L.kd, Mc, L.kd, L.rw, L.kd, L.cout, nullptr, bf.R, L.cout,
         L.rb, &L.rbn, nullptr, 0, false);
    res = bf.R;
  }
  // tcn 1x1 conv + BN2 + residual + ReLU
  gemm(st, bf.H, L.cout, Mc, L.cout, L.tw, L.cout, L.cout, nullptr, Xout, L.cout,
       L.tb, &L.bn2, res, L.cout, true);
}

// TemporalConvNetBlock: cat(relu(conv5), relu(conv3(relu(conv3)))) -> 1x1 -> BN -> ReLU
static void tcn_block(hipStream_t st, const TcnW& T, const h16* Zin, Bufs& bf,
                      h16* outH, float* outF, int ldcOut) {
  long tot = (long)NN * FC;
  int blks = (int)((tot + 255) / 256);
  im2col_k<<<blks, 256, 0, st>>>(Zin, bf.I5, BB, SS, FC, 5);
  gemm(st, bf.I5, FC * 5, NN, FC * 5, T.w5, FC * 5, FC, nullptr, bf.Cat, 2 * FC,
       T.b5, nullptr, nullptr, 0, true);
  im2col_k<<<blks, 256, 0, st>>>(Zin, bf.I3, BB, SS, FC, 3);
  gemm(st, bf.I3, FC * 3, NN, FC * 3, T.w3a, FC * 3, FC, nullptr, bf.Btmp, FC,
       T.b3a, nullptr, nullptr, 0, true);
  im2col_k<<<blks, 256, 0, st>>>(bf.Btmp, bf.I3, BB, SS, FC, 3);
  gemm(st, bf.I3, FC * 3, NN, FC * 3, T.w3b, FC * 3, FC, nullptr, bf.Cat + FC, 2 * FC,
       T.b3b, nullptr, nullptr, 0, true);
  gemm(st, bf.Cat, 2 * FC, NN, 2 * FC, T.w1, 2 * FC, FC, outF, outH,
       outF ? ldcOut : FC, T.b1, &T.bn, nullptr, 0, true);
}

static void run_stream(hipStream_t st, const float* x, const float* Aadj, const float* Apool,
                       const ModelW& MW, int V, int P, int set, Bufs& bf, float* outBase) {
  // data_bn (eps = 0.1) -> X0 f16 rows (n*V+v), K padded to 32
  long rows0 = (long)NN * V;
  long tot0 = rows0 * KPAD;
  data_bn_k<<<(int)((tot0 + 255) / 256), 256, 0, st>>>(
      x, bf.X0, rows0, V, CIN, KPAD, MW.dbn.g, MW.dbn.b, MW.dbn.m, MW.dbn.v);

  for (int n0 = 0; n0 < NN; n0 += NC) {
    int nc = (NN - n0 < NC) ? (NN - n0) : NC;
    const h16* Xin = bf.X0 + (long)n0 * V * KPAD;
    run_layer(st, MW.gcn[0], Aadj, nc, V, Xin,   bf, bf.Xa);
    run_layer(st, MW.gcn[1], Aadj, nc, V, bf.Xa, bf, bf.Xb);
    run_layer(st, MW.gcn[2], Aadj, nc, V, bf.Xb, bf, bf.Xa);
    long tp = (long)nc * P * MW.gcn[2].cout;
    part_pool_k<<<(int)((tp + 255) / 256), 256, 0, st>>>(
        bf.Xa, bf.Xb, nc, V, MW.gcn[2].cout, P, set);
    run_layer(st, MW.pool[0], Apool, nc, P, bf.Xb, bf, bf.Xa);
    run_layer(st, MW.pool[1], Apool, nc, P, bf.Xa, bf, bf.Xb);
    long tgm = (long)nc * FC;
    gmax_k<<<(int)((tgm + 255) / 256), 256, 0, st>>>(
        bf.Xb, bf.Zg + (long)n0 * FC, nc, P, FC);
  }

  // conv4: FC x FC + BN + ReLU
  gemm(st, bf.Zg, FC, NN, FC, MW.c4w, FC, FC, nullptr, bf.Z1, FC,
       MW.c4b, &MW.c4bn, nullptr, 0, true);
  // two TCN blocks; last one writes f32 straight into the strided output
  tcn_block(st, MW.t1, bf.Z1, bf, bf.Z2, nullptr, 0);
  tcn_block(st, MW.t2, bf.Z2, bf, nullptr, outBase, 3 * FC);
}

// ---------------------------------------------------------------------------
// Entry point
// ---------------------------------------------------------------------------
extern "C" void kernel_launch(void* const* d_in, const int* in_sizes, int n_in,
                              void* d_out, int out_size, void* d_ws, size_t ws_size,
                              hipStream_t stream) {
  (void)in_sizes; (void)n_in; (void)out_size;

  Bump ws{(char*)d_ws, 0, ws_size};

  // Param leaf indices: 7 top-level arrays, then hand_params, then body_params
  int idx = 7;
  ModelIdx hIdx{}, bIdx{};
  walkModel(idx, hIdx);
  walkModel(idx, bIdx);

  // Convert all weights to f16 (deterministic allocation order per call)
  ModelW handW = buildModel(ws, stream, d_in, hIdx);
  ModelW bodyW = buildModel(ws, stream, d_in, bIdx);

  // Scratch buffers (sized for hand graph, V=21; reused across streams/chunks)
  Bufs bf{};
  size_t actEl = (size_t)NC * 21 * 256;                    // per-chunk activation tile
  bf.X0   = (h16*)ws.get((size_t)NN * 21 * KPAD * sizeof(h16));
  bf.Xa   = (h16*)ws.get(actEl * sizeof(h16));
  bf.Xb   = (h16*)ws.get(actEl * sizeof(h16));
  bf.H    = (h16*)ws.get(actEl * sizeof(h16));
  bf.R    = (h16*)ws.get(actEl * sizeof(h16));
  bf.Y1   = (h16*)ws.get((size_t)NC * 21 * (KK * 256) * sizeof(h16));
  bf.Zg   = (h16*)ws.get((size_t)NN * FC * sizeof(h16));
  bf.Z1   = (h16*)ws.get((size_t)NN * FC * sizeof(h16));
  bf.Z2   = (h16*)ws.get((size_t)NN * FC * sizeof(h16));
  bf.I5   = (h16*)ws.get((size_t)NN * FC * 5 * sizeof(h16));
  bf.I3   = (h16*)ws.get((size_t)NN * FC * 3 * sizeof(h16));
  bf.Btmp = (h16*)ws.get((size_t)NN * FC * sizeof(h16));
  bf.Cat  = (h16*)ws.get((size_t)NN * FC * 2 * sizeof(h16));

  const float* right = (const float*)d_in[0];
  const float* left  = (const float*)d_in[1];
  const float* body  = (const float*)d_in[2];
  const float* Ah    = (const float*)d_in[3];
  const float* Ab    = (const float*)d_in[4];
  const float* Aph   = (const float*)d_in[5];
  const float* Apb   = (const float*)d_in[6];
  float* out = (float*)d_out;

  // Output layout (B, S, 3*FC): [left | right | body]
  run_stream(stream, left,  Ah, Aph, handW, 21, 6, 0, bf, out + 0);
  run_stream(stream, right, Ah, Aph, handW, 21, 6, 0, bf, out + FC);
  run_stream(stream, body,  Ab, Apb, bodyW, 13, 5, 1, bf, out + 2 * FC);
}
